// MultiSpaceAttention_35656818492076
// MI455X (gfx1250) — matrix operand order repaired and checked by reading, not compiled, not run on hardware
//
#include <hip/hip_runtime.h>

// ---------------------------------------------------------------------------
// MultiSpaceAttention for MI455X (gfx1250): bf16 WMMA everywhere, attn tensor
// touched the minimum number of times (~1.07 GB HBM traffic dominates).
// All GEMM k-loops are software-pipelined (double-buffered fragments) so the
// 10 global_load_b128 of step k+1 are in flight while step k's 4 WMMAs run.
// ---------------------------------------------------------------------------

typedef unsigned short u16;
typedef unsigned int   u32;
typedef __attribute__((ext_vector_type(4)))  float  f32x4;
typedef __attribute__((ext_vector_type(8)))  float  f32x8;
typedef __attribute__((ext_vector_type(8)))  unsigned short u16x8;
typedef __attribute__((ext_vector_type(4)))  unsigned short u16x4;
typedef __attribute__((ext_vector_type(16))) __bf16 bf16x16;

#define D_MODEL 512
#define N_HEADS 8
#define DEPTH   64
#define SEQ     1024
#define BGDIM   8                    // B*G = 2*4
#define NROWS   (BGDIM * SEQ)        // 8192
#define NHB     (BGDIM * N_HEADS)    // 64 head-batches

// ---- helpers ---------------------------------------------------------------

__device__ __forceinline__ u16 f2bf(float x) {
  u32 u = __builtin_bit_cast(u32, x);
  u32 r = u + 0x7FFFu + ((u >> 16) & 1u);   // round-to-nearest-even
  return (u16)(r >> 16);
}

// A fragment (16x32 bf16): p is pre-offset by row*ld + ks*32 + hi8.
// Lane's 16 K-values are two contiguous 16B chunks at +0 and +16 elements.
__device__ __forceinline__ bf16x16 fragA16(const u16* p) {
  union { bf16x16 v; u16x8 h[2]; } r;
  r.h[0] = *(const u16x8*)(p);
  r.h[1] = *(const u16x8*)(p + 16);
  return r.v;
}

// B fragment (32x16 bf16) from a B^T (N-major, K-contiguous) buffer:
// p is pre-offset by (n-col)*ld + ks*32 + hi8; two 16B chunks at +0 and +16.
__device__ __forceinline__ bf16x16 fragB16(const u16* p) {
  union { bf16x16 v; u16x8 h[2]; } r;
  r.h[0] = *(const u16x8*)(p);
  r.h[1] = *(const u16x8*)(p + 16);
  return r.v;
}

// raw f32 A-chunk (attn matrix): 4x f32x4 loads, converted separately so the
// loads can be double-buffered and the cvt VALU co-executes with WMMA.
struct araw { f32x4 a, b, c, d; };
__device__ __forceinline__ araw loadA32(const float* p) {   // p pre-offset (+hi8)
  araw r;
  r.a = *(const f32x4*)(p);
  r.b = *(const f32x4*)(p + 4);
  r.c = *(const f32x4*)(p + 16);
  r.d = *(const f32x4*)(p + 20);
  return r;
}
__device__ __forceinline__ bf16x16 cvtA(const araw& s) {
  union { bf16x16 v; u16 u[16]; } r;
#pragma unroll
  for (int i = 0; i < 4; ++i) {
    r.u[i]      = f2bf(s.a[i]);
    r.u[4 + i]  = f2bf(s.b[i]);
    r.u[8 + i]  = f2bf(s.c[i]);
    r.u[12 + i] = f2bf(s.d[i]);
  }
  return r.v;
}

__device__ __forceinline__ f32x8 wmma_bf16(bf16x16 a, bf16x16 b, f32x8 c) {
  // (neg_a, A, neg_b, B, c_mod, C, reuse_a, reuse_b)
  return __builtin_amdgcn_wmma_f32_16x16x32_bf16(false, a, false, b,
                                                 (short)0, c, false, false);
}

// ---- stage 0: precision staging -------------------------------------------

__global__ void k_cvt_bf16(const float* __restrict__ src, u16* __restrict__ dst) {
  int i = (blockIdx.x * blockDim.x + threadIdx.x) * 4;
  f32x4 v = *(const f32x4*)(src + i);
  u16x4 o;
#pragma unroll
  for (int k = 0; k < 4; ++k) o[k] = f2bf(v[k]);
  *(u16x4*)(dst + i) = o;
}

// src [rows][cols] f32  ->  dst [cols][rows] bf16
__global__ void k_transpose_bf16(const float* __restrict__ src, u16* __restrict__ dst,
                                 int rows, int cols) {
  int i = blockIdx.x * blockDim.x + threadIdx.x;
  int r = i / cols, c = i % cols;
  dst[c * rows + r] = f2bf(src[i]);
}

// ---- stage 1: fused QKV projection + head split ----------------------------

__global__ void k_qkv(const u16* __restrict__ qb, const u16* __restrict__ wxT,
                      const float* __restrict__ wx_b,
                      u16* __restrict__ qhb, u16* __restrict__ khb,
                      u16* __restrict__ vtb) {
  const int lane = threadIdx.x & 31;
  const int wave = blockIdx.x * (blockDim.x >> 5) + (threadIdx.x >> 5);
  const int NG = (3 * D_MODEL) / 64;                 // 24 column groups
  const int mt = wave / NG, ng = wave % NG;
  const int m0 = mt * 16, n0 = ng * 64;
  const int lm = lane & 15, lh = lane >> 4;
  const int hi8 = lh * 8, kh16 = lh * 16;
  constexpr int BSTR = 16 * D_MODEL;
  constexpr int KS = D_MODEL / 32;                   // 16

  const u16* ap = qb + (m0 + lm) * D_MODEL + hi8;
  const u16* bp = wxT + (n0 + lm) * D_MODEL + hi8;

  f32x8 c[4] = {};
  bf16x16 a  = fragA16(ap);
  bf16x16 b0 = fragB16(bp);
  bf16x16 b1 = fragB16(bp + BSTR);
  bf16x16 b2 = fragB16(bp + 2 * BSTR);
  bf16x16 b3 = fragB16(bp + 3 * BSTR);
#pragma unroll 5
  for (int ks = 0; ks < KS - 1; ++ks) {
    const int o = (ks + 1) * 32;
    bf16x16 an  = fragA16(ap + o);
    bf16x16 bn0 = fragB16(bp + o);
    bf16x16 bn1 = fragB16(bp + BSTR + o);
    bf16x16 bn2 = fragB16(bp + 2 * BSTR + o);
    bf16x16 bn3 = fragB16(bp + 3 * BSTR + o);
    c[0] = wmma_bf16(a, b0, c[0]);
    c[1] = wmma_bf16(a, b1, c[1]);
    c[2] = wmma_bf16(a, b2, c[2]);
    c[3] = wmma_bf16(a, b3, c[3]);
    a = an; b0 = bn0; b1 = bn1; b2 = bn2; b3 = bn3;
  }
  c[0] = wmma_bf16(a, b0, c[0]);
  c[1] = wmma_bf16(a, b1, c[1]);
  c[2] = wmma_bf16(a, b2, c[2]);
  c[3] = wmma_bf16(a, b3, c[3]);

#pragma unroll
  for (int j = 0; j < 4; ++j) {
    const int ncol = n0 + j * 16 + lm;
    const int which = ncol >> 9;          // 0=q 1=k 2=v (uniform per wave)
    const int d = ncol & 511;
    const int h = d >> 6, dd = d & 63;
    const float bias = wx_b[ncol];
#pragma unroll
    for (int r = 0; r < 8; ++r) {
      const int rowM = m0 + r + lh * 8;
      const int bg = rowM >> 10, t = rowM & 1023;
      const int head = bg * N_HEADS + h;
      const u16 val = f2bf(c[j][r] + bias);
      if (which == 0)      qhb[(head * SEQ + t) * DEPTH + dd] = val;
      else if (which == 1) khb[(head * SEQ + t) * DEPTH + dd] = val;
      else                 vtb[head * DEPTH * SEQ + dd * SEQ + t] = val;
    }
  }
}

// ---- stage 2: logits = Q K^T / sqrt(64) + mask*(-1e9) ----------------------
// K = 64 -> only 2 k-steps: hoist all 20 fragment loads, then 8 WMMAs.

__global__ void k_logits(const u16* __restrict__ qhb, const u16* __restrict__ khb,
                         const int* __restrict__ mask, float* __restrict__ attn) {
  const int lane = threadIdx.x & 31;
  const int wave = blockIdx.x * (blockDim.x >> 5) + (threadIdx.x >> 5);
  const int ng = wave & 15, mt = (wave >> 4) & 63, head = wave >> 10;
  const int m0 = mt * 16, n0 = ng * 64;
  const int lm = lane & 15, lh = lane >> 4;
  const int hi8 = lh * 8, kh16 = lh * 16;
  constexpr int BSTR = 16 * DEPTH;

  const u16* ap = qhb + head * SEQ * DEPTH + (m0 + lm) * DEPTH + hi8;
  const u16* bp = khb + head * SEQ * DEPTH + (n0 + lm) * DEPTH + hi8;

  bf16x16 a0 = fragA16(ap),      a1 = fragA16(ap + 32);
  bf16x16 b00 = fragB16(bp),              b01 = fragB16(bp + BSTR);
  bf16x16 b02 = fragB16(bp + 2 * BSTR),   b03 = fragB16(bp + 3 * BSTR);
  bf16x16 b10 = fragB16(bp + 32),         b11 = fragB16(bp + BSTR + 32);
  bf16x16 b12 = fragB16(bp + 2 * BSTR + 32), b13 = fragB16(bp + 3 * BSTR + 32);

  f32x8 c[4] = {};
  c[0] = wmma_bf16(a0, b00, c[0]);
  c[1] = wmma_bf16(a0, b01, c[1]);
  c[2] = wmma_bf16(a0, b02, c[2]);
  c[3] = wmma_bf16(a0, b03, c[3]);
  c[0] = wmma_bf16(a1, b10, c[0]);
  c[1] = wmma_bf16(a1, b11, c[1]);
  c[2] = wmma_bf16(a1, b12, c[2]);
  c[3] = wmma_bf16(a1, b13, c[3]);

  float* out = attn + (size_t)head * SEQ * SEQ;
  const float scale = 0.125f;                         // 1/sqrt(64)
#pragma unroll
  for (int j = 0; j < 4; ++j) {
    const int kcol = n0 + j * 16 + lm;
#pragma unroll
    for (int r = 0; r < 8; ++r) {
      const int qrow = m0 + r + lh * 8;
      out[qrow * SEQ + kcol] =
          c[j][r] * scale + (float)mask[qrow * SEQ + kcol] * (-1e9f);
    }
  }
}

// ---- stage 3: row softmax in place -----------------------------------------

__global__ void k_softmax(float* __restrict__ attn) {
  float* p = attn + (size_t)blockIdx.x * SEQ;
  const int tid = threadIdx.x;
  float x[4];
#pragma unroll
  for (int i = 0; i < 4; ++i) x[i] = p[tid + 256 * i];

  float m = fmaxf(fmaxf(x[0], x[1]), fmaxf(x[2], x[3]));
#pragma unroll
  for (int off = 16; off > 0; off >>= 1) m = fmaxf(m, __shfl_xor(m, off, 32));
  __shared__ float red[8];
  if ((tid & 31) == 0) red[tid >> 5] = m;
  __syncthreads();
#pragma unroll
  for (int w = 0; w < 8; ++w) m = fmaxf(m, red[w]);

  float s = 0.0f;
#pragma unroll
  for (int i = 0; i < 4; ++i) { x[i] = __expf(x[i] - m); s += x[i]; }
#pragma unroll
  for (int off = 16; off > 0; off >>= 1) s += __shfl_xor(s, off, 32);
  __syncthreads();
  if ((tid & 31) == 0) red[tid >> 5] = s;
  __syncthreads();
  s = 0.0f;
#pragma unroll
  for (int w = 0; w < 8; ++w) s += red[w];

  const float inv = 1.0f / s;
#pragma unroll
  for (int i = 0; i < 4; ++i) p[tid + 256 * i] = x[i] * inv;
}

// ---- stage 4: out_h = attn @ V  (A: f32->bf16 on load, B: V^T bf16) --------

__global__ void k_av(const float* __restrict__ attn, const u16* __restrict__ vtb,
                     u16* __restrict__ oh) {
  const int lane = threadIdx.x & 31;
  const int wave = blockIdx.x * (blockDim.x >> 5) + (threadIdx.x >> 5);
  const int mt = wave & 63, head = wave >> 6;
  const int m0 = mt * 16;
  const int lm = lane & 15, lh = lane >> 4;
  const int hi8 = lh * 8, kh16 = lh * 16;
  constexpr int BSTR = 16 * SEQ;
  constexpr int KS = SEQ / 32;                        // 32

  const float* ap = attn + (size_t)head * SEQ * SEQ + (size_t)(m0 + lm) * SEQ + hi8;
  const u16*   bp = vtb + head * DEPTH * SEQ + lm * SEQ + hi8;

  f32x8 c[4] = {};
  araw    ar = loadA32(ap);
  bf16x16 b0 = fragB16(bp);
  bf16x16 b1 = fragB16(bp + BSTR);
  bf16x16 b2 = fragB16(bp + 2 * BSTR);
  bf16x16 b3 = fragB16(bp + 3 * BSTR);
#pragma unroll 4
  for (int ks = 0; ks < KS - 1; ++ks) {
    const int o = (ks + 1) * 32;
    // long-range prefetch of the attn stream (~1KB ahead, clamped in-row)
    const int pfo = (ks + 8 < KS ? ks + 8 : KS - 1) * 32;
    __builtin_prefetch(ap + pfo, 0, 1);               // global_prefetch_b8
    araw    an  = loadA32(ap + o);
    bf16x16 bn0 = fragB16(bp + o);
    bf16x16 bn1 = fragB16(bp + BSTR + o);
    bf16x16 bn2 = fragB16(bp + 2 * BSTR + o);
    bf16x16 bn3 = fragB16(bp + 3 * BSTR + o);
    bf16x16 a = cvtA(ar);                             // VALU co-executes w/ WMMA
    c[0] = wmma_bf16(a, b0, c[0]);
    c[1] = wmma_bf16(a, b1, c[1]);
    c[2] = wmma_bf16(a, b2, c[2]);
    c[3] = wmma_bf16(a, b3, c[3]);
    ar = an; b0 = bn0; b1 = bn1; b2 = bn2; b3 = bn3;
  }
  {
    bf16x16 a = cvtA(ar);
    c[0] = wmma_bf16(a, b0, c[0]);
    c[1] = wmma_bf16(a, b1, c[1]);
    c[2] = wmma_bf16(a, b2, c[2]);
    c[3] = wmma_bf16(a, b3, c[3]);
  }

  u16* orow = oh + head * SEQ * DEPTH;
#pragma unroll
  for (int j = 0; j < 4; ++j) {
    const int dd = j * 16 + lm;
#pragma unroll
    for (int r = 0; r < 8; ++r) {
      const int t = m0 + r + lh * 8;
      orow[t * DEPTH + dd] = f2bf(c[j][r]);
    }
  }
}

// ---- stage 5: output projection --------------------------------------------

__global__ void k_oproj(const u16* __restrict__ oh, const u16* __restrict__ woT,
                        const float* __restrict__ wo_b, float* __restrict__ out) {
  const int lane = threadIdx.x & 31;
  const int wave = blockIdx.x * (blockDim.x >> 5) + (threadIdx.x >> 5);
  const int ng = wave & 7, mt = wave >> 3;
  const int m0 = mt * 16, n0 = ng * 64;
  const int lm = lane & 15, lh = lane >> 4;
  const int hi8 = lh * 8, kh16 = lh * 16;
  const int rm = m0 + lm;
  const int bg = rm >> 10, t = rm & 1023;
  constexpr int BSTR = 16 * D_MODEL;
  constexpr int KS = D_MODEL / 32;                    // 16

  // A K-index walks per-head 64-chunks of oh: offset(ks) = (ks>>1)*SEQ*DEPTH + (ks&1)*32
  const u16* ap = oh + (bg * N_HEADS * SEQ + t) * DEPTH + hi8;
  const u16* bp = woT + (n0 + lm) * D_MODEL + hi8;
#define AOFF(ks) (((ks) >> 1) * (SEQ * DEPTH) + ((ks) & 1) * 32)

  f32x8 c[4] = {};
  bf16x16 a  = fragA16(ap);
  bf16x16 b0 = fragB16(bp);
  bf16x16 b1 = fragB16(bp + BSTR);
  bf16x16 b2 = fragB16(bp + 2 * BSTR);
  bf16x16 b3 = fragB16(bp + 3 * BSTR);
#pragma unroll 5
  for (int ks = 0; ks < KS - 1; ++ks) {
    const int oa = AOFF(ks + 1);
    const int ob = (ks + 1) * 32;
    bf16x16 an  = fragA16(ap + oa);
    bf16x16 bn0 = fragB16(bp + ob);
    bf16x16 bn1 = fragB16(bp + BSTR + ob);
    bf16x16 bn2 = fragB16(bp + 2 * BSTR + ob);
    bf16x16 bn3 = fragB16(bp + 3 * BSTR + ob);
    c[0] = wmma_bf16(a, b0, c[0]);
    c[1] = wmma_bf16(a, b1, c[1]);
    c[2] = wmma_bf16(a, b2, c[2]);
    c[3] = wmma_bf16(a, b3, c[3]);
    a = an; b0 = bn0; b1 = bn1; b2 = bn2; b3 = bn3;
  }
  c[0] = wmma_bf16(a, b0, c[0]);
  c[1] = wmma_bf16(a, b1, c[1]);
  c[2] = wmma_bf16(a, b2, c[2]);
  c[3] = wmma_bf16(a, b3, c[3]);
#undef AOFF

#pragma unroll
  for (int j = 0; j < 4; ++j) {
    const int ncol = n0 + j * 16 + lm;
    const float bias = wo_b[ncol];
#pragma unroll
    for (int r = 0; r < 8; ++r) {
      const int rowM = m0 + r + lh * 8;
      out[rowM * D_MODEL + ncol] = c[j][r] + bias;
    }
  }
}

// ---------------------------------------------------------------------------

extern "C" void kernel_launch(void* const* d_in, const int* in_sizes, int n_in,
                              void* d_out, int out_size, void* d_ws, size_t ws_size,
                              hipStream_t stream) {
  // inputs: v(0), k(1), q(2), mask(3), wx_w(4), wx_b(5), wo_w(6), wo_b(7)
  const float* q    = (const float*)d_in[2];
  const int*   mask = (const int*)d_in[3];
  const float* wx_w = (const float*)d_in[4];
  const float* wx_b = (const float*)d_in[5];
  const float* wo_w = (const float*)d_in[6];
  const float* wo_b = (const float*)d_in[7];

  float* out  = (float*)d_out;
  float* attn = out + (size_t)NROWS * D_MODEL;        // outputs: (out, attn)

  // workspace layout (bf16 elements)
  u16* ws  = (u16*)d_ws;
  u16* qb  = ws;                    // [8192][512]        4194304
  u16* wxT = qb  + 4194304;         // [1536][512]         786432
  u16* woT = wxT + 786432;          // [512][512]          262144
  u16* qhb = woT + 262144;          // [64][1024][64]     4194304
  u16* khb = qhb + 4194304;         // [64][1024][64]     4194304
  u16* vtb = khb + 4194304;         // [64][64][1024]     4194304
  u16* oh  = vtb + 4194304;         // [64][1024][64]     4194304

  k_cvt_bf16      <<<4096, 256, 0, stream>>>(q, qb);
  k_transpose_bf16<<<3072, 256, 0, stream>>>(wx_w, wxT, 512, 1536);
  k_transpose_bf16<<<1024, 256, 0, stream>>>(wo_w, woT, 512, 512);
  k_qkv           <<<1536, 256, 0, stream>>>(qb, wxT, wx_b, qhb, khb, vtb);   // 512*24 waves
  k_logits        <<<8192, 256, 0, stream>>>(qhb, khb, mask, attn);           // 64*64*16 waves
  k_softmax       <<<65536, 256, 0, stream>>>(attn);                          // one row / block
  k_av            <<<512, 256, 0, stream>>>(attn, vtb, oh);                   // 64*64 waves
  k_oproj         <<<512, 256, 0, stream>>>(oh, woT, wo_b, out);              // 512*8 waves
}
